// RSHBlock_20486994001926
// MI455X (gfx1250) — matrix operand order repaired and checked by reading, not compiled, not run on hardware
//
#include <hip/hip_runtime.h>
#include <hip/hip_bf16.h>
#include <math.h>

// MI455X / gfx1250: all matrix contractions on v_wmma_f32_16x16x32_f16
// (wave32, 8-wave workgroups). Weights pre-packed to f16 fragment order so
// every B operand is a contiguous 32B-per-lane load; contiguous A panels are
// staged to LDS via the CDNA5 async copy unit (ASYNCcnt).

typedef __attribute__((ext_vector_type(16))) _Float16 v16h;
typedef __attribute__((ext_vector_type(8)))  float    v8f;

#define NB 2
#define HWSZ 65536
#define CL 96
#define CG 96
#define CD 192
#define TOK 64

// ---- WMMA fragment helpers (CDNA5 16x16x32 f16 layouts) --------------------

// A fragment 16x32 (MxK): element (m,k) at src + m*rs + k*cs
template <typename T>
__device__ __forceinline__ v16h ldA(const T* __restrict__ src, int rs, int cs) {
  const int lane = threadIdx.x & 31;
  const T* p = src + (lane & 15) * rs + ((lane >> 4) << 3) * cs;
  v16h f;
#pragma unroll
  for (int j = 0; j < 8; ++j) {
    f[j]     = (_Float16)p[j * cs];
    f[j + 8] = (_Float16)p[(16 + j) * cs];
  }
  return f;
}

// B fragment 32x16 (KxN): element (k,n) at src + k*ks + n*ns
template <typename T>
__device__ __forceinline__ v16h ldB(const T* __restrict__ src, int ks, int ns) {
  const int lane = threadIdx.x & 31;
  const T* p = src + (lane & 15) * ns + ((lane >> 4) << 3) * ks;
  v16h f;
#pragma unroll
  for (int j = 0; j < 8; ++j) {
    f[j]     = (_Float16)p[j * ks];
    f[j + 8] = (_Float16)p[(16 + j) * ks];
  }
  return f;
}

// B fragment from pre-packed weights: contiguous 16 halves per lane
__device__ __forceinline__ v16h ldBp(const _Float16* __restrict__ packed,
                                     int tile) {
  const _Float16* p = packed + (size_t)tile * 512 + (threadIdx.x & 31) * 16;
  v16h f;
#pragma unroll
  for (int j = 0; j < 16; ++j) f[j] = p[j];
  return f;
}

__device__ __forceinline__ v8f wmma32(v16h a, v16h b, v8f c) {
  return __builtin_amdgcn_wmma_f32_16x16x32_f16(false, a, false, b, (short)0, c,
                                                false, false);
}

// CDNA5 async global->LDS copy (16B per issue), tracked by ASYNCcnt.
__device__ __forceinline__ void async_b128(const _Float16* g, _Float16* l) {
  unsigned lds = (unsigned)(size_t)l;
  unsigned long long ga = (unsigned long long)(size_t)g;
  asm volatile("global_load_async_to_lds_b128 %0, %1, off" ::"v"(lds), "v"(ga)
               : "memory");
}
__device__ __forceinline__ void async_wait() {
  asm volatile("s_wait_asynccnt 0x0" ::: "memory");
}

__device__ __forceinline__ float lrelu(float x) { return x >= 0.f ? x : 0.2f * x; }

// Pack W (N x K logical, element (n,k) = src[n*rs + k*cs]) into f16
// fragment-order tiles: dst[tile*512 + lane*16 + j], tile = nt*ktiles + kt.
__global__ void k_pack(const float* __restrict__ src, _Float16* __restrict__ dst,
                       int ktiles, int rs, int cs) {
  int tile = blockIdx.x;
  int nt = tile / ktiles, kt = tile % ktiles;
  for (int e = threadIdx.x; e < 512; e += 256) {
    int lane = e >> 4, j = e & 15;
    int k = (kt << 5) + ((lane >> 4) << 3) + (j & 7) + ((j >> 3) << 4);
    int n = (nt << 4) + (lane & 15);
    dst[(size_t)tile * 512 + e] = (_Float16)src[n * rs + k * cs];
  }
}

// ---- Local branch: 3x3 convs as implicit GEMM ------------------------------
__global__ __launch_bounds__(256) void k_conv(
    const float* __restrict__ xin, const _Float16* __restrict__ pw,
    float* __restrict__ outbuf, float* __restrict__ gsum, float* __restrict__ gsq,
    const float* __restrict__ mean, const float* __restrict__ rstd,
    const float* __restrict__ gamma, const float* __restrict__ beta, int mode) {
  __shared__ _Float16 shA[128 * 96];
  __shared__ float rsum[16], rsq[16];
  const int tid = threadIdx.x;
  const int b = blockIdx.z;
  const int oc0 = blockIdx.y * 16;
  const int p0 = blockIdx.x * 128;
  if (tid < 16) { rsum[tid] = 0.f; rsq[tid] = 0.f; }
  const int wv = tid >> 5, lane = tid & 31;
  v8f acc = {};
  for (int tap = 0; tap < 9; ++tap) {
    const int ky = tap / 3 - 1, kx = tap % 3 - 1;
    __syncthreads();
    for (int e = tid; e < 128 * 96; e += 256) {
      int i = e / 96, c = e % 96;
      int p = p0 + i;
      int h = (p >> 8) + ky, w = (p & 255) + kx;
      float v = 0.f;
      if (h >= 0 && h < 256 && w >= 0 && w < 256) {
        int q = h * 256 + w;
        if (mode == 0) {
          v = xin[(b * CD + c) * HWSZ + q];
        } else {
          float t = xin[(b * CL + c) * HWSZ + q];
          int bc = b * CL + c;
          v = lrelu((t - mean[bc]) * rstd[bc] * gamma[c] + beta[c]);
        }
      }
      shA[e] = (_Float16)v;
    }
    __syncthreads();
#pragma unroll
    for (int kk = 0; kk < 96; kk += 32) {
      v16h a = ldA(shA + wv * 16 * 96 + kk, 96, 1);
      v16h bf = ldBp(pw + tap * 9216, blockIdx.y * 3 + (kk >> 5));
      acc = wmma32(a, bf, acc);
    }
  }
  const int n16 = lane & 15, mo = (lane >> 4) << 3;
#pragma unroll
  for (int i = 0; i < 8; ++i) {
    int pix = p0 + wv * 16 + mo + i;
    float v = acc[i];
    if (mode == 1) v = lrelu(v);
    outbuf[(b * CL + oc0 + n16) * HWSZ + pix] = v;
    atomicAdd(&rsum[n16], v);
    if (mode == 0) atomicAdd(&rsq[n16], v * v);
  }
  __syncthreads();
  if (tid < 16) {
    atomicAdd(&gsum[b * CL + oc0 + tid], rsum[tid]);
    if (mode == 0) atomicAdd(&gsq[b * CL + oc0 + tid], rsq[tid]);
  }
}

__global__ void k_stats(const float* gsum, const float* gsq, float* mean,
                        float* rstd) {
  int i = threadIdx.x;
  if (i < NB * CL) {
    float m = gsum[i] / (float)HWSZ;
    float v = gsq[i] / (float)HWSZ - m * m;
    mean[i] = m;
    rstd[i] = rsqrtf(v + 1e-5f);
  }
}

__global__ __launch_bounds__(128) void k_se(const float* sesum, const float* w1,
                                            const float* w2, float* sey) {
  __shared__ float mv[96], hv[24];
  int b = blockIdx.x, t = threadIdx.x;
  if (t < 96) mv[t] = sesum[b * 96 + t] / (float)HWSZ;
  __syncthreads();
  if (t < 24) {
    float s = 0.f;
    for (int c = 0; c < 96; ++c) s += mv[c] * w1[t * 96 + c];
    hv[t] = s > 0.f ? s : 0.f;
  }
  __syncthreads();
  if (t < 96) {
    float s = 0.f;
    for (int j = 0; j < 24; ++j) s += hv[j] * w2[t * 24 + j];
    sey[b * 96 + t] = 1.f / (1.f + expf(-s));
  }
}

__global__ void k_lxfinal(const float* lx2, const float* sey, const float* x,
                          float* lxf) {
  size_t idx = (size_t)blockIdx.x * 256 + threadIdx.x;  // b*HW + p
  int b = (int)(idx >> 16), p = (int)(idx & 65535);
  for (int c = 0; c < 96; ++c) {
    float v = lx2[(b * CL + c) * HWSZ + p] * sey[b * 96 + c] +
              x[((size_t)b * CD + c) * HWSZ + p];
    lxf[idx * 96 + c] = v;
  }
}

__global__ void k_gxext(const float* x, float* gx) {
  size_t idx = (size_t)blockIdx.x * 256 + threadIdx.x;
  int b = (int)(idx >> 16), p = (int)(idx & 65535);
  for (int c = 0; c < 96; ++c)
    gx[idx * 96 + c] = x[((size_t)b * CD + 96 + c) * HWSZ + p];
}

// ---- Swin ------------------------------------------------------------------
__global__ void k_ln(const float* __restrict__ gx, const float* __restrict__ g,
                     const float* __restrict__ bp, _Float16* __restrict__ ln) {
  size_t idx = (size_t)blockIdx.x * 256 + threadIdx.x;
  const float* r = gx + idx * 96;
  float s = 0.f, sq = 0.f;
  for (int c = 0; c < 96; ++c) { float v = r[c]; s += v; sq += v * v; }
  float m = s / 96.f;
  float rs = rsqrtf(sq / 96.f - m * m + 1e-5f);
  _Float16* o = ln + idx * 96;
  for (int c = 0; c < 96; ++c) o[c] = (_Float16)((r[c] - m) * rs * g[c] + bp[c]);
}

// One workgroup per (window, batch): qkv GEMM, 3-head attention, fused proj.
__global__ __launch_bounds__(256) void k_swin_attn(
    const _Float16* __restrict__ ln, float* __restrict__ gx,
    const _Float16* __restrict__ pqkv, const float* __restrict__ qkvb,
    const float* __restrict__ rel, const _Float16* __restrict__ pproj,
    const float* __restrict__ projb, int shifted) {
  __shared__ _Float16 shq[TOK * 96];
  __shared__ _Float16 shkT[96 * TOK];  // K stored transposed: [c][tok]
  __shared__ _Float16 shv[TOK * 96];
  __shared__ float shs[TOK * TOK];
  __shared__ _Float16 sho[TOK * 32];
  __shared__ int rowoff[TOK];
  const int tid = threadIdx.x, wv = tid >> 5, lane = tid & 31;
  const int b = blockIdx.y;
  const int wy = blockIdx.x >> 5, wx = blockIdx.x & 31;
  const int sh = shifted ? 4 : 0;
  if (tid < TOK) {
    int ty = tid >> 3, tx = tid & 7;
    int h = ((wy << 3) + ty + sh) & 255;
    int w = ((wx << 3) + tx + sh) & 255;
    rowoff[tid] = (b * HWSZ + h * 256 + w) * 96;
  }
  __syncthreads();
  const int n16 = lane & 15, mo = (lane >> 4) << 3;
  // QKV: 64x96 @ 96x288 (72 tiles over 8 waves)
  for (int t = wv; t < 72; t += 8) {
    int mi = t & 3, ni = t >> 2;
    v8f acc = {};
#pragma unroll
    for (int kk = 0; kk < 96; kk += 32) {
      v16h a;
      {
        int m = (mi << 4) + (lane & 15);
        const _Float16* p = ln + rowoff[m] + kk + ((lane >> 4) << 3);
#pragma unroll
        for (int j = 0; j < 8; ++j) { a[j] = p[j]; a[j + 8] = p[16 + j]; }
      }
      v16h bf = ldBp(pqkv, ni * 3 + (kk >> 5));
      acc = wmma32(a, bf, acc);
    }
    int col = (ni << 4) + n16;
    float bias = qkvb[col];
#pragma unroll
    for (int i = 0; i < 8; ++i) {
      int tok = (mi << 4) + mo + i;
      float v = acc[i] + bias;
      if (col < 96) shq[tok * 96 + col] = (_Float16)(v * 0.17677669529663687f);
      else if (col < 192) shkT[(col - 96) * 64 + tok] = (_Float16)v;
      else shv[tok * 96 + (col - 192)] = (_Float16)v;
    }
  }
  __syncthreads();
  v8f pacc[3] = {};
  for (int hd = 0; hd < 3; ++hd) {
    // sim = q @ k^T + rel bias (+ shift mask)
    for (int t = wv; t < 16; t += 8) {
      int mi = t & 3, ni = t >> 2;
      v16h a = ldA(shq + (mi * 16) * 96 + hd * 32, 96, 1);
      v16h bf = ldB(shkT + (hd * 32) * 64 + ni * 16, 64, 1);
      v8f acc = {};
      acc = wmma32(a, bf, acc);
#pragma unroll
      for (int i = 0; i < 8; ++i) {
        int pr = (mi << 4) + mo + i, qc = (ni << 4) + n16;
        int py = pr >> 3, px = pr & 7, qy = qc >> 3, qx = qc & 7;
        float v = acc[i] + rel[hd * 225 + (py - qy + 7) * 15 + (px - qx + 7)];
        if (shifted && ((wy == 31 && ((py < 4) != (qy < 4))) ||
                        (wx == 31 && ((px < 4) != (qx < 4)))))
          v = -1e30f;
        shs[pr * 64 + qc] = v;
      }
    }
    __syncthreads();
    if (tid < 64) {  // softmax, one row per lane
      float* row = shs + tid * 64;
      float mx = -1e30f;
      for (int j = 0; j < 64; ++j) mx = fmaxf(mx, row[j]);
      float s = 0.f;
      for (int j = 0; j < 64; ++j) { float e = expf(row[j] - mx); row[j] = e; s += e; }
      float inv = 1.f / s;
      for (int j = 0; j < 64; ++j) row[j] *= inv;
    }
    __syncthreads();
    // out_head = probs @ v_head (8 tiles == 8 waves)
    {
      int mi = wv & 3, ni = (wv >> 2) & 1;
      v8f acc = {};
#pragma unroll
      for (int kk = 0; kk < 64; kk += 32) {
        v16h a = ldA(shs + (mi * 16) * 64 + kk, 64, 1);
        v16h bf = ldB(shv + kk * 96 + hd * 32 + ni * 16, 96, 1);
        acc = wmma32(a, bf, acc);
      }
#pragma unroll
      for (int i = 0; i < 8; ++i)
        sho[((mi << 4) + mo + i) * 32 + (ni << 4) + n16] = (_Float16)acc[i];
    }
    __syncthreads();
    // proj partial accumulation over heads (K=32 per head)
    for (int t = wv, j = 0; t < 24; t += 8, ++j) {
      int mi = t & 3, ni = t >> 2;
      v16h a = ldA(sho + (mi * 16) * 32, 32, 1);
      v16h bf = ldBp(pproj, ni * 3 + hd);
      pacc[j] = wmma32(a, bf, pacc[j]);
    }
    __syncthreads();
  }
  // proj epilogue: + bias + residual, in place into gx
  for (int t = wv, j = 0; t < 24; t += 8, ++j) {
    int mi = t & 3, ni = t >> 2;
    int col = (ni << 4) + n16;
    float bias = projb[col];
#pragma unroll
    for (int i = 0; i < 8; ++i) {
      int tok = (mi << 4) + mo + i;
      float* dst = gx + rowoff[tok] + col;
      *dst = pacc[j][i] + bias + *dst;
    }
  }
}

__global__ __launch_bounds__(256) void k_mlp(
    const _Float16* __restrict__ ln, float* __restrict__ gx,
    const _Float16* __restrict__ pw1, const float* __restrict__ b1,
    const _Float16* __restrict__ pw2, const float* __restrict__ b2) {
  __shared__ _Float16 shx[64 * 96];
  __shared__ _Float16 shh[64 * 384];
  const int tid = threadIdx.x, wv = tid >> 5, lane = tid & 31;
  const int n16 = lane & 15, mo = (lane >> 4) << 3;
  const size_t p0 = (size_t)blockIdx.x * 64;
  // async-stage the 64x96 f16 input panel (reused by 24 output tiles)
  {
    const _Float16* gsrc = ln + p0 * 96;
    for (int r = tid; r < 768; r += 256) async_b128(gsrc + r * 8, shx + r * 8);
    async_wait();
  }
  __syncthreads();
  for (int t = wv; t < 96; t += 8) {
    int mi = t & 3, ni = t >> 2;
    v8f acc = {};
#pragma unroll
    for (int kk = 0; kk < 96; kk += 32) {
      v16h a = ldA(shx + (mi * 16) * 96 + kk, 96, 1);
      v16h bf = ldBp(pw1, ni * 3 + (kk >> 5));
      acc = wmma32(a, bf, acc);
    }
    int col = (ni << 4) + n16;
    float bias = b1[col];
#pragma unroll
    for (int i = 0; i < 8; ++i) {
      float v = acc[i] + bias;
      v = 0.5f * v * (1.f + erff(v * 0.7071067811865475f));
      shh[((mi << 4) + mo + i) * 384 + col] = (_Float16)v;
    }
  }
  __syncthreads();
  for (int t = wv; t < 24; t += 8) {
    int mi = t & 3, ni = t >> 2;
    v8f acc = {};
#pragma unroll
    for (int kk = 0; kk < 384; kk += 32) {
      __builtin_prefetch(pw2 + ((size_t)(ni * 12 + (kk >> 5)) + 1) * 512, 0, 0);
      v16h a = ldA(shh + (mi * 16) * 384 + kk, 384, 1);
      v16h bf = ldBp(pw2, ni * 12 + (kk >> 5));
      acc = wmma32(a, bf, acc);
    }
    int col = (ni << 4) + n16;
    float bias = b2[col];
#pragma unroll
    for (int i = 0; i < 8; ++i)
      gx[(p0 + (mi << 4) + mo + i) * 96 + col] += acc[i] + bias;
  }
}

// ---- Channel attention -----------------------------------------------------
__global__ void k_cat(const float* lxf, const float* gx, _Float16* catb) {
  size_t idx = (size_t)blockIdx.x * 256 + threadIdx.x;
  for (int c = 0; c < 96; ++c) {
    catb[idx * 192 + c] = (_Float16)lxf[idx * 96 + c];
    catb[idx * 192 + 96 + c] = (_Float16)gx[idx * 96 + c];
  }
}

__global__ __launch_bounds__(256) void k_ca_qkv(const _Float16* __restrict__ catb,
                                                const _Float16* __restrict__ pw,
                                                _Float16* __restrict__ qkv) {
  __shared__ _Float16 shx[64 * 192];
  const int tid = threadIdx.x, wv = tid >> 5, lane = tid & 31;
  const int n16 = lane & 15, mo = (lane >> 4) << 3;
  const size_t p0 = (size_t)blockIdx.x * 64;
  const int oc0 = blockIdx.y * 64;
  {
    const _Float16* gsrc = catb + p0 * 192;
    for (int r = tid; r < 1536; r += 256) async_b128(gsrc + r * 8, shx + r * 8);
    async_wait();
  }
  __syncthreads();
  for (int t = wv; t < 16; t += 8) {
    int mi = t & 3, ni = t >> 2;
    v8f acc = {};
#pragma unroll
    for (int kk = 0; kk < 192; kk += 32) {
      v16h a = ldA(shx + (mi * 16) * 192 + kk, 192, 1);
      v16h bf = ldBp(pw, ((oc0 >> 4) + ni) * 6 + (kk >> 5));
      acc = wmma32(a, bf, acc);
    }
    int col = oc0 + (ni << 4) + n16;
#pragma unroll
    for (int i = 0; i < 8; ++i)
      qkv[(p0 + (mi << 4) + mo + i) * 576 + col] = (_Float16)acc[i];
  }
}

// depthwise 3x3; writes pixel-major (for B operands) and channel-major (for A)
__global__ void k_ca_dw(const _Float16* __restrict__ qkv,
                        const float* __restrict__ dww,
                        _Float16* __restrict__ dwo, _Float16* __restrict__ dwoC) {
  size_t idx = (size_t)blockIdx.x * 256 + threadIdx.x;  // b*HW*576
  int c = (int)(idx % 576);
  size_t bp = idx / 576;
  int p = (int)(bp & 65535), b = (int)(bp >> 16);
  int h = p >> 8, w = p & 255;
  float acc = 0.f;
#pragma unroll
  for (int ky = -1; ky <= 1; ++ky)
#pragma unroll
    for (int kx = -1; kx <= 1; ++kx) {
      int hh = h + ky, w2 = w + kx;
      if (hh >= 0 && hh < 256 && w2 >= 0 && w2 < 256)
        acc += (float)qkv[((size_t)b * HWSZ + hh * 256 + w2) * 576 + c] *
               dww[c * 9 + (ky + 1) * 3 + (kx + 1)];
    }
  _Float16 hv = (_Float16)acc;
  dwo[idx] = hv;
  dwoC[((size_t)(b * 576 + c)) * HWSZ + p] = hv;
}

__global__ __launch_bounds__(256) void k_ca_norm(const _Float16* __restrict__ dwoC,
                                                 float* __restrict__ nrm) {
  __shared__ float red[256];
  int b = blockIdx.x / 384, c = blockIdx.x % 384;
  const _Float16* src = dwoC + ((size_t)(b * 576 + c)) * HWSZ;
  float s = 0.f;
  for (int p = threadIdx.x; p < HWSZ; p += 256) {
    float v = (float)src[p];
    s += v * v;
  }
  red[threadIdx.x] = s;
  __syncthreads();
  for (int o = 128; o > 0; o >>= 1) {
    if (threadIdx.x < o) red[threadIdx.x] += red[threadIdx.x + o];
    __syncthreads();
  }
  if (threadIdx.x == 0) nrm[blockIdx.x] = fmaxf(sqrtf(red[0]), 1e-12f);
}

// Q.K^T with split-K over HW, f32 atomic merge (32x32 per (b,head)).
// A from channel-major (contiguous K), B from pixel-major (contiguous N).
__global__ __launch_bounds__(256) void k_ca_attn(const _Float16* __restrict__ dwo,
                                                 const _Float16* __restrict__ dwoC,
                                                 float* __restrict__ araw) {
  const int tid = threadIdx.x, wv = tid >> 5, lane = tid & 31;
  const int n16 = lane & 15, mo = (lane >> 4) << 3;
  const int hd = blockIdx.y, b = blockIdx.z;
  const int mi = (wv >> 1) & 1, ni = wv & 1, ks = wv >> 2;
  const size_t pb = (size_t)blockIdx.x * 4096 + (size_t)ks * 2048;
  const _Float16* Abase =
      dwoC + ((size_t)(b * 576 + hd * 32 + mi * 16)) * HWSZ + pb;
  v8f acc = {};
  for (int kk = 0; kk < 2048; kk += 32) {
    v16h a = ldA(Abase + kk, HWSZ, 1);
    v16h bf = ldB(dwo + ((size_t)b * HWSZ + pb + kk) * 576 + 192 + hd * 32 +
                      ni * 16,
                  576, 1);
    acc = wmma32(a, bf, acc);
  }
  float* dst = araw + (size_t)(b * 6 + hd) * 1024;
#pragma unroll
  for (int i = 0; i < 8; ++i)
    atomicAdd(&dst[(mi * 16 + mo + i) * 32 + ni * 16 + n16], acc[i]);
}

__global__ void k_ca_soft(const float* araw, const float* nrm, float* probs) {
  int bh = blockIdx.x, b = bh / 6, hd = bh % 6;
  int i = threadIdx.x;
  const float* r = araw + (bh * 32 + i) * 32;
  float qn = nrm[b * 384 + hd * 32 + i];
  float row[32];
  float mx = -1e30f;
  for (int j = 0; j < 32; ++j) {
    float kn = nrm[b * 384 + 192 + hd * 32 + j];
    float v = r[j] / (qn * kn) * 0.17677669529663687f;
    row[j] = v;
    mx = fmaxf(mx, v);
  }
  float s = 0.f;
  for (int j = 0; j < 32; ++j) { row[j] = expf(row[j] - mx); s += row[j]; }
  float inv = 1.f / s;
  float* o = probs + (bh * 32 + i) * 32;
  for (int j = 0; j < 32; ++j) o[j] = row[j] * inv;
}

// out^T = V^T @ attn^T : M=16 pixels (streaming, contiguous A), N=32 channels
__global__ __launch_bounds__(256) void k_ca_out(const float* __restrict__ probs,
                                                const _Float16* __restrict__ dwo,
                                                _Float16* __restrict__ vout) {
  const int tid = threadIdx.x, wv = tid >> 5, lane = tid & 31;
  const int n16 = lane & 15, mo = (lane >> 4) << 3;
  const int hd = blockIdx.y, b = blockIdx.z;
  const int bh = b * 6 + hd;
  const size_t pg = (size_t)blockIdx.x * 128 + wv * 16;
  v16h a = ldA(dwo + ((size_t)b * HWSZ + pg) * 576 + 384 + hd * 32, 576, 1);
#pragma unroll
  for (int ni = 0; ni < 2; ++ni) {
    v16h bf = ldB(probs + ((size_t)bh * 32 + ni * 16) * 32, 1, 32);
    v8f acc = {};
    acc = wmma32(a, bf, acc);
#pragma unroll
    for (int i = 0; i < 8; ++i)
      vout[((size_t)b * HWSZ + pg + mo + i) * 192 + hd * 32 + ni * 16 + n16] =
          (_Float16)acc[i];
  }
}

// final 1x1 proj + x_cat + original residual -> NCHW fp32 out
__global__ __launch_bounds__(256) void k_ca_proj(
    const _Float16* __restrict__ vout, const _Float16* __restrict__ pw,
    const float* __restrict__ lxf, const float* __restrict__ gx,
    const float* __restrict__ x, float* __restrict__ out) {
  __shared__ _Float16 shx[64 * 192];
  const int tid = threadIdx.x, wv = tid >> 5, lane = tid & 31;
  const int n16 = lane & 15, mo = (lane >> 4) << 3;
  const size_t p0 = (size_t)blockIdx.x * 64;
  const int oc0 = blockIdx.y * 64;
  {
    const _Float16* gsrc = vout + p0 * 192;
    for (int r = tid; r < 1536; r += 256) async_b128(gsrc + r * 8, shx + r * 8);
    async_wait();
  }
  __syncthreads();
  for (int t = wv; t < 16; t += 8) {
    int mi = t & 3, ni = t >> 2;
    v8f acc = {};
#pragma unroll
    for (int kk = 0; kk < 192; kk += 32) {
      v16h a = ldA(shx + (mi * 16) * 192 + kk, 192, 1);
      v16h bf = ldBp(pw, ((oc0 >> 4) + ni) * 6 + (kk >> 5));
      acc = wmma32(a, bf, acc);
    }
    int col = oc0 + (ni << 4) + n16;
#pragma unroll
    for (int i = 0; i < 8; ++i) {
      size_t pp = p0 + (mi << 4) + mo + i;
      int b = (int)(pp >> 16), p = (int)(pp & 65535);
      float xc = (col < 96) ? lxf[pp * 96 + col] : gx[pp * 96 + (col - 96)];
      float rs = x[((size_t)b * CD + col) * HWSZ + p];
      out[((size_t)b * CD + col) * HWSZ + p] = acc[i] + xc + rs;
    }
  }
}

// ---------------------------------------------------------------------------
extern "C" void kernel_launch(void* const* d_in, const int* in_sizes, int n_in,
                              void* d_out, int out_size, void* d_ws,
                              size_t ws_size, hipStream_t stream) {
  const float* x = (const float*)d_in[0];
  const float* rw1 = (const float*)d_in[1];
  const float* rw2 = (const float*)d_in[2];
  const float* rgam = (const float*)d_in[3];
  const float* rbet = (const float*)d_in[4];
  const float* sew1 = (const float*)d_in[5];
  const float* sew2 = (const float*)d_in[6];
  const float* sp[2][13];
  for (int s = 0; s < 2; ++s)
    for (int j = 0; j < 13; ++j) sp[s][j] = (const float*)d_in[7 + s * 13 + j];
  const float* caq = (const float*)d_in[33];
  const float* cad = (const float*)d_in[34];
  const float* cap = (const float*)d_in[35];
  float* out = (float*)d_out;

  char* ws = (char*)d_ws;
  float* sm = (float*)ws;
  float *g_sum = sm, *g_sq = sm + 192, *g_mean = sm + 384, *g_rstd = sm + 576;
  float *g_sesum = sm + 768, *g_sey = sm + 960;
  float *g_araw = sm + 1152, *g_probs = sm + 13440, *g_nrm = sm + 25728;
  size_t off = 131072;
  float* lx1 = (float*)(ws + off); off += 50331648ull;
  float* lx2 = (float*)(ws + off); off += 50331648ull;
  float* gx = (float*)(ws + off); off += 50331648ull;
  float* lxf = (float*)(ws + off); off += 50331648ull;
  _Float16* lnb = (_Float16*)(ws + off); off += 25165824ull;
  _Float16* catb = (_Float16*)(ws + off); off += 50331648ull;
  _Float16* qkvb = (_Float16*)(ws + off); off += 150994944ull;
  _Float16* dwob = (_Float16*)(ws + off); off += 150994944ull;
  _Float16* dwoCb = (_Float16*)(ws + off); off += 150994944ull;
  _Float16* voutb = (_Float16*)(ws + off); off += 50331648ull;
  // packed weights (f16 fragment order)
  _Float16* pconv1 = (_Float16*)(ws + off); off += 165888ull;
  _Float16* pconv2 = (_Float16*)(ws + off); off += 165888ull;
  _Float16* pqkv = (_Float16*)(ws + off); off += 110592ull;   // 2 x 27648 halves
  _Float16* pproj = (_Float16*)(ws + off); off += 36864ull;   // 2 x 9216
  _Float16* pw1 = (_Float16*)(ws + off); off += 147456ull;    // 2 x 36864
  _Float16* pw2 = (_Float16*)(ws + off); off += 147456ull;    // 2 x 36864
  _Float16* pcaq = (_Float16*)(ws + off); off += 221184ull;   // 110592 halves
  _Float16* pcap = (_Float16*)(ws + off); off += 73728ull;    // 36864 halves

  hipMemsetAsync(ws, 0, 131072, stream);  // zero stat/attn accumulators

  // weight pre-pack (tiny; L2-resident afterwards)
  for (int tap = 0; tap < 9; ++tap) {
    k_pack<<<18, 256, 0, stream>>>(rw1 + tap, pconv1 + tap * 9216, 3, 864, 9);
    k_pack<<<18, 256, 0, stream>>>(rw2 + tap, pconv2 + tap * 9216, 3, 864, 9);
  }
  for (int s = 0; s < 2; ++s) {
    k_pack<<<54, 256, 0, stream>>>(sp[s][2], pqkv + s * 27648, 3, 96, 1);
    k_pack<<<18, 256, 0, stream>>>(sp[s][5], pproj + s * 9216, 3, 96, 1);
    k_pack<<<72, 256, 0, stream>>>(sp[s][9], pw1 + s * 36864, 3, 96, 1);
    k_pack<<<72, 256, 0, stream>>>(sp[s][11], pw2 + s * 36864, 12, 384, 1);
  }
  k_pack<<<216, 256, 0, stream>>>(caq, pcaq, 6, 192, 1);
  k_pack<<<72, 256, 0, stream>>>(cap, pcap, 6, 192, 1);

  // local branch
  k_conv<<<dim3(512, 6, 2), 256, 0, stream>>>(x, pconv1, lx1, g_sum, g_sq,
                                              g_mean, g_rstd, rgam, rbet, 0);
  k_stats<<<1, 192, 0, stream>>>(g_sum, g_sq, g_mean, g_rstd);
  k_conv<<<dim3(512, 6, 2), 256, 0, stream>>>(lx1, pconv2, lx2, g_sesum, g_sq,
                                              g_mean, g_rstd, rgam, rbet, 1);
  k_se<<<2, 128, 0, stream>>>(g_sesum, sew1, sew2, g_sey);
  k_lxfinal<<<512, 256, 0, stream>>>(lx2, g_sey, x, lxf);

  // global branch: 2 swin blocks (second shifted)
  k_gxext<<<512, 256, 0, stream>>>(x, gx);
  for (int s = 0; s < 2; ++s) {
    k_ln<<<512, 256, 0, stream>>>(gx, sp[s][0], sp[s][1], lnb);
    k_swin_attn<<<dim3(1024, 2), 256, 0, stream>>>(
        lnb, gx, pqkv + s * 27648, sp[s][3], sp[s][4], pproj + s * 9216,
        sp[s][6], s);
    k_ln<<<512, 256, 0, stream>>>(gx, sp[s][7], sp[s][8], lnb);
    k_mlp<<<2048, 256, 0, stream>>>(lnb, gx, pw1 + s * 36864, sp[s][10],
                                    pw2 + s * 36864, sp[s][12]);
  }

  // channel attention + residuals
  k_cat<<<512, 256, 0, stream>>>(lxf, gx, catb);
  k_ca_qkv<<<dim3(2048, 9), 256, 0, stream>>>(catb, pcaq, qkvb);
  k_ca_dw<<<294912, 256, 0, stream>>>(qkvb, cad, dwob, dwoCb);
  k_ca_norm<<<768, 256, 0, stream>>>(dwoCb, g_nrm);
  k_ca_attn<<<dim3(16, 6, 2), 256, 0, stream>>>(dwob, dwoCb, g_araw);
  k_ca_soft<<<12, 32, 0, stream>>>(g_araw, g_nrm, g_probs);
  k_ca_out<<<dim3(512, 6, 2), 256, 0, stream>>>(g_probs, dwob, voutb);
  k_ca_proj<<<dim3(2048, 3), 256, 0, stream>>>(voutb, pcap, lxf, gx, x, out);

  (void)in_sizes; (void)n_in; (void)out_size; (void)ws_size;
}
